// DeformableTransformer_31473520345581
// MI455X (gfx1250) — matrix operand order repaired and checked by reading, not compiled, not run on hardware
//
#include <hip/hip_runtime.h>
#include <hip/hip_bf16.h>
#include <math.h>

// ---------------- problem constants ----------------
#define D_MODEL 256
#define NHEADS  8
#define NLEV    4
#define NPTS    4
#define DHEAD   32
#define DFF     1024
#define NLAYERS 6
#define LQ      12240
#define BATCH   2
#define M_TOK   (BATCH * LQ)   // 24480 rows, multiple of 32

// level geometry (all levels are square: h == w)
__constant__ int c_hw[4]    = {96, 48, 24, 12};
__constant__ int c_start[4] = {0, 9216, 11520, 12096};
__constant__ int c_size[4]  = {9216, 2304, 576, 144};

typedef __attribute__((ext_vector_type(16))) __bf16 v16bf;
typedef __attribute__((ext_vector_type(8)))  __bf16 bf16x8;
typedef __attribute__((ext_vector_type(8)))  float  v8f;

// ---------------- init: build x and pos from src/pos pyramids ----------------
__global__ void k_init_x_pos(const float* __restrict__ s0, const float* __restrict__ s1,
                             const float* __restrict__ s2, const float* __restrict__ s3,
                             const float* __restrict__ p0, const float* __restrict__ p1,
                             const float* __restrict__ p2, const float* __restrict__ p3,
                             const float* __restrict__ lvl,
                             float* __restrict__ x, float* __restrict__ pos)
{
    int idx = blockIdx.x * blockDim.x + threadIdx.x;      // over M_TOK*256
    if (idx >= M_TOK * D_MODEL) return;
    int d = idx & (D_MODEL - 1);
    int t = idx >> 8;                                     // b*LQ + q
    int b = t / LQ;
    int q = t - b * LQ;
    int l, f;
    if (q < 9216)       { l = 0; f = q; }
    else if (q < 11520) { l = 1; f = q - 9216; }
    else if (q < 12096) { l = 2; f = q - 11520; }
    else                { l = 3; f = q - 12096; }
    const float* sp; const float* pp;
    if (l == 0)      { sp = s0; pp = p0; }
    else if (l == 1) { sp = s1; pp = p1; }
    else if (l == 2) { sp = s2; pp = p2; }
    else             { sp = s3; pp = p3; }
    int hw = c_size[l];
    size_t si = ((size_t)b * D_MODEL + d) * hw + f;       // src layout (B, D, h*w)
    x[idx]   = sp[si];
    pos[idx] = pp[si] + lvl[l * D_MODEL + d];
}

// ---------------- per-layer prep: q_bf = bf16(x+pos), x_bf = bf16(x) -------
__global__ void k_prep(const float* __restrict__ x, const float* __restrict__ pos,
                       __bf16* __restrict__ q_bf, __bf16* __restrict__ x_bf)
{
    int idx = blockIdx.x * blockDim.x + threadIdx.x;
    if (idx >= M_TOK * D_MODEL) return;
    float xv = x[idx];
    q_bf[idx] = (__bf16)(xv + pos[idx]);
    x_bf[idx] = (__bf16)xv;
}

// ---------------- weight pack: f32 (K,N) row-major -> bf16 WMMA B fragments -
// Fragment (kt, nt): 32 lanes x 16 bf16 contiguous per lane.
// lane<16 -> column n = nt*16+lane, K = kt*32 + 0..15 (ascending)
// lane>=16 -> same column, K = kt*32 + 16..31
__global__ void k_pack(const float* __restrict__ W, __bf16* __restrict__ out,
                       int K, int N)
{
    int idx = blockIdx.x * blockDim.x + threadIdx.x;
    if (idx >= K * N) return;
    int k = idx / N, n = idx - k * N;
    int kt = k >> 5, kin = k & 31;
    int nt = n >> 4, nl = n & 15;
    int lane = (kin < 16) ? nl : (16 + nl);
    int slot = kin & 15;
    size_t o = (((size_t)kt * (N >> 4) + nt) * 32 + lane) * 16 + slot;
    out[o] = (__bf16)W[idx];
}

// ---------------- WMMA GEMM: C[M,N] = A[M,K](bf16) * Wp + bias -------------
// Register-blocked: one wave computes a 32x64 tile (2 M-frags x 4 N-frags),
// 8 x v_wmma_f32_16x16x32_bf16 per K-step; A frags reused 4x, B frags 2x.
__global__ void k_gemm(const __bf16* __restrict__ A, const __bf16* __restrict__ Wp,
                       const float* __restrict__ bias,
                       float* __restrict__ Cf, __bf16* __restrict__ Cb,
                       int Mr, int N, int K, int relu)
{
    int wave = (blockIdx.x * blockDim.x + threadIdx.x) >> 5;
    int lane = threadIdx.x & 31;
    int nBlks = N >> 6;                     // 64-wide N blocks
    int total = (Mr >> 5) * nBlks;          // 32-tall M blocks
    if (wave >= total) return;
    int mBlk = wave / nBlks;
    int nBlk = wave - mBlk * nBlks;
    int half = lane >> 4;
    int l15  = lane & 15;

    const __bf16* aRow0 = A + (size_t)(mBlk * 32 + l15) * K + half * 16;
    const __bf16* aRow1 = aRow0 + (size_t)16 * K;
    // B fragment base for nTile = nBlk*4 + j is bBase + j*512
    const __bf16* bBase = Wp + ((size_t)(nBlk * 4) * 32 + lane) * 16;
    size_t bStride = (size_t)(N >> 4) * 512;            // elems per kt step

    v8f acc[2][4];
#pragma unroll
    for (int mi = 0; mi < 2; ++mi)
#pragma unroll
        for (int j = 0; j < 4; ++j) acc[mi][j] = (v8f){};

    int kts = K >> 5;
    for (int kt = 0; kt < kts; ++kt) {
        const __bf16* a0 = aRow0 + kt * 32;
        const __bf16* a1 = aRow1 + kt * 32;
        const __bf16* bp = bBase + (size_t)kt * bStride;

        // prefetch next K-step fragments (speculative, OOB-safe)
        __builtin_prefetch(a0 + 32, 0, 1);
        __builtin_prefetch(a1 + 32, 0, 1);
        __builtin_prefetch(bp + bStride, 0, 1);

        bf16x8 a0lo = *(const bf16x8*)a0;
        bf16x8 a0hi = *(const bf16x8*)(a0 + 8);
        bf16x8 a1lo = *(const bf16x8*)a1;
        bf16x8 a1hi = *(const bf16x8*)(a1 + 8);
        v16bf af[2];
        af[0] = __builtin_shufflevector(a0lo, a0hi, 0,1,2,3,4,5,6,7,8,9,10,11,12,13,14,15);
        af[1] = __builtin_shufflevector(a1lo, a1hi, 0,1,2,3,4,5,6,7,8,9,10,11,12,13,14,15);

        v16bf bf[4];
#pragma unroll
        for (int j = 0; j < 4; ++j) {
            const __bf16* bj = bp + (size_t)j * 512;
            bf16x8 blo = *(const bf16x8*)bj;
            bf16x8 bhi = *(const bf16x8*)(bj + 8);
            bf[j] = __builtin_shufflevector(blo, bhi, 0,1,2,3,4,5,6,7,8,9,10,11,12,13,14,15);
        }

#pragma unroll
        for (int mi = 0; mi < 2; ++mi)
#pragma unroll
            for (int j = 0; j < 4; ++j)
                acc[mi][j] = __builtin_amdgcn_wmma_f32_16x16x32_bf16(
                    false, af[mi], false, bf[j], (short)0, acc[mi][j], false, false);
    }

#pragma unroll
    for (int j = 0; j < 4; ++j) {
        int col = nBlk * 64 + j * 16 + l15;
        float bv = bias ? bias[col] : 0.0f;
#pragma unroll
        for (int mi = 0; mi < 2; ++mi) {
            int rowBase = mBlk * 32 + mi * 16 + half * 8;
#pragma unroll
            for (int r = 0; r < 8; ++r) {
                float v = acc[mi][j][r] + bv;
                if (relu) v = fmaxf(v, 0.0f);
                size_t o = (size_t)(rowBase + r) * N + col;
                if (Cf) Cf[o] = v;
                if (Cb) Cb[o] = (__bf16)v;
            }
        }
    }
}

// ---------------- deformable attention sampling ----------------------------
// One wave per (b, q, head); lane = channel within head (DHEAD == 32).
__global__ void k_sample(const float* __restrict__ off, const float* __restrict__ attn,
                         const float* __restrict__ val, __bf16* __restrict__ samp)
{
    int gw = (blockIdx.x * blockDim.x + threadIdx.x) >> 5;
    int lane = threadIdx.x & 31;
    if (gw >= M_TOK * NHEADS) return;
    int h = gw & (NHEADS - 1);
    int t = gw >> 3;                  // b*LQ + q
    int b = t / LQ;
    int q = t - b * LQ;

    int ql, qf;
    if (q < 9216)       { ql = 0; qf = q; }
    else if (q < 11520) { ql = 1; qf = q - 9216; }
    else if (q < 12096) { ql = 2; qf = q - 11520; }
    else                { ql = 3; qf = q - 12096; }
    int qwid = c_hw[ql];
    int qr = qf / qwid, qc = qf - qr * qwid;
    float refx = (qc + 0.5f) / (float)qwid;
    float refy = (qr + 0.5f) / (float)qwid;

    // softmax over NL*NP = 16 logits
    const float* ap = attn + (size_t)t * (NHEADS * 16) + h * 16;
    float lg[16];
    float mx = -1e30f;
#pragma unroll
    for (int j = 0; j < 16; ++j) { lg[j] = ap[j]; mx = fmaxf(mx, lg[j]); }
    float se = 0.0f;
#pragma unroll
    for (int j = 0; j < 16; ++j) { lg[j] = __expf(lg[j] - mx); se += lg[j]; }
    float inv = 1.0f / se;

    const float* op = off + (size_t)t * D_MODEL + h * (NLEV * NPTS * 2);

    float acc = 0.0f;
#pragma unroll
    for (int l = 0; l < NLEV; ++l) {
        int HW = c_hw[l];
        int st = c_start[l];
        const float* vbase = val + (((size_t)b * LQ + st) * NHEADS + h) * DHEAD + lane;
        float fHW = (float)HW;
#pragma unroll
        for (int p = 0; p < NPTS; ++p) {
            float aw = lg[l * 4 + p] * inv;
            float ox = op[(l * 4 + p) * 2 + 0];
            float oy = op[(l * 4 + p) * 2 + 1];
            float xx = (refx + ox / fHW) * fHW - 0.5f;
            float yy = (refy + oy / fHW) * fHW - 0.5f;
            float x0f = floorf(xx), y0f = floorf(yy);
            float lx = xx - x0f, ly = yy - y0f;
            int x0 = (int)x0f, y0 = (int)y0f;
            float w00 = (1.0f - lx) * (1.0f - ly);
            float w10 = lx * (1.0f - ly);
            float w01 = (1.0f - lx) * ly;
            float w11 = lx * ly;
            float s = 0.0f;
#pragma unroll
            for (int cr = 0; cr < 4; ++cr) {
                int xi = x0 + (cr & 1);
                int yi = y0 + (cr >> 1);
                float wgt = (cr == 0) ? w00 : (cr == 1) ? w10 : (cr == 2) ? w01 : w11;
                float valid = (xi >= 0 && xi < HW && yi >= 0 && yi < HW) ? 1.0f : 0.0f;
                int cx = min(max(xi, 0), HW - 1);
                int cy = min(max(yi, 0), HW - 1);
                float vv = vbase[(size_t)(cy * HW + cx) * (NHEADS * DHEAD)];
                s += wgt * valid * vv;
            }
            acc += aw * s;
        }
    }
    samp[(size_t)t * D_MODEL + h * DHEAD + lane] = (__bf16)acc;
}

// ---------------- residual + LayerNorm (wave per row) ----------------------
__global__ void k_addln(const float* __restrict__ x, const float* __restrict__ r,
                        const float* __restrict__ g, const float* __restrict__ bta,
                        float* __restrict__ xo, __bf16* __restrict__ xbf)
{
    int gw = (blockIdx.x * blockDim.x + threadIdx.x) >> 5;
    int lane = threadIdx.x & 31;
    if (gw >= M_TOK) return;
    size_t base = (size_t)gw * D_MODEL;
    float v[8];
    float s = 0.0f;
#pragma unroll
    for (int k = 0; k < 8; ++k) {
        int c = lane + k * 32;
        v[k] = x[base + c] + r[base + c];
        s += v[k];
    }
#pragma unroll
    for (int o = 16; o > 0; o >>= 1) s += __shfl_xor(s, o, 32);
    float mu = s * (1.0f / D_MODEL);
    float s2 = 0.0f;
#pragma unroll
    for (int k = 0; k < 8; ++k) { float d = v[k] - mu; s2 += d * d; }
#pragma unroll
    for (int o = 16; o > 0; o >>= 1) s2 += __shfl_xor(s2, o, 32);
    float rinv = rsqrtf(s2 * (1.0f / D_MODEL) + 1e-5f);
#pragma unroll
    for (int k = 0; k < 8; ++k) {
        int c = lane + k * 32;
        float o = (v[k] - mu) * rinv * g[c] + bta[c];
        xo[base + c] = o;
        if (xbf) xbf[base + c] = (__bf16)o;
    }
}

// ---------------- host driver ----------------------------------------------
static inline int blocks_for(long long threads) { return (int)((threads + 255) / 256); }

extern "C" void kernel_launch(void* const* d_in, const int* in_sizes, int n_in,
                              void* d_out, int out_size, void* d_ws, size_t ws_size,
                              hipStream_t stream)
{
    (void)in_sizes; (void)n_in; (void)out_size; (void)ws_size;
    // dict order: src0,pos0,src1,pos1,src2,pos2,src3,pos3, level_embed,
    // W_off,b_off, W_attn,b_attn, W_val,b_val, W_out,b_out, ln1_g,ln1_b,
    // W_ffn1,b_ffn1, W_ffn2,b_ffn2, ln2_g,ln2_b
    const float* src[4] = {(const float*)d_in[0], (const float*)d_in[2],
                           (const float*)d_in[4], (const float*)d_in[6]};
    const float* pin[4] = {(const float*)d_in[1], (const float*)d_in[3],
                           (const float*)d_in[5], (const float*)d_in[7]};
    const float* lvl    = (const float*)d_in[8];
    const float* W_off  = (const float*)d_in[9];
    const float* b_off  = (const float*)d_in[10];
    const float* W_attn = (const float*)d_in[11];
    const float* b_attn = (const float*)d_in[12];
    const float* W_val  = (const float*)d_in[13];
    const float* b_val  = (const float*)d_in[14];
    const float* W_out  = (const float*)d_in[15];
    const float* b_out  = (const float*)d_in[16];
    const float* ln1_g  = (const float*)d_in[17];
    const float* ln1_b  = (const float*)d_in[18];
    const float* W_f1   = (const float*)d_in[19];
    const float* b_f1   = (const float*)d_in[20];
    const float* W_f2   = (const float*)d_in[21];
    const float* b_f2   = (const float*)d_in[22];
    const float* ln2_g  = (const float*)d_in[23];
    const float* ln2_b  = (const float*)d_in[24];

    // workspace carve-up (256B aligned); aliased buffers noted
    char* base = (char*)d_ws;
    size_t cur = 0;
    auto carve = [&](size_t bytes) -> char* {
        char* p = base + cur;
        cur += (bytes + 255) & ~(size_t)255;
        return p;
    };
    const size_t MD = (size_t)M_TOK * D_MODEL;
    float*  xbuf   = (float*) carve(MD * 4);
    float*  posbuf = (float*) carve(MD * 4);
    __bf16* q_bf   = (__bf16*)carve(MD * 2);          // aliased: samp_bf
    __bf16* x_bf   = (__bf16*)carve(MD * 2);          // reused after LN1 for FFN input
    float*  offb   = (float*) carve(MD * 4);          // aliased: proj
    float*  attnb  = (float*) carve((size_t)M_TOK * 128 * 4);
    float*  valb   = (float*) carve(MD * 4);          // aliased: ffn2 out
    __bf16* ffn1b  = (__bf16*)carve((size_t)M_TOK * DFF * 2);
    __bf16* wp_off = (__bf16*)carve((size_t)256 * 256 * 2);
    __bf16* wp_att = (__bf16*)carve((size_t)256 * 128 * 2);
    __bf16* wp_val = (__bf16*)carve((size_t)256 * 256 * 2);
    __bf16* wp_out = (__bf16*)carve((size_t)256 * 256 * 2);
    __bf16* wp_f1  = (__bf16*)carve((size_t)256 * DFF * 2);
    __bf16* wp_f2  = (__bf16*)carve((size_t)DFF * 256 * 2);
    __bf16* samp   = q_bf;      // alias (q_bf dead once off/attn GEMMs are done)
    float*  proj   = offb;      // alias (off dead once sampling is done)
    float*  ffn2   = valb;      // alias (val dead once sampling is done)

    // build x, pos
    k_init_x_pos<<<blocks_for(MD), 256, 0, stream>>>(
        src[0], src[1], src[2], src[3], pin[0], pin[1], pin[2], pin[3],
        lvl, xbuf, posbuf);

    for (int li = 0; li < NLAYERS; ++li) {
        // pack this layer's weights to bf16 fragments
        k_pack<<<blocks_for(256 * 256), 256, 0, stream>>>(W_off  + (size_t)li * 256 * 256, wp_off, 256, 256);
        k_pack<<<blocks_for(256 * 128), 256, 0, stream>>>(W_attn + (size_t)li * 256 * 128, wp_att, 256, 128);
        k_pack<<<blocks_for(256 * 256), 256, 0, stream>>>(W_val  + (size_t)li * 256 * 256, wp_val, 256, 256);
        k_pack<<<blocks_for(256 * 256), 256, 0, stream>>>(W_out  + (size_t)li * 256 * 256, wp_out, 256, 256);
        k_pack<<<blocks_for(256 * DFF), 256, 0, stream>>>(W_f1   + (size_t)li * 256 * DFF, wp_f1, 256, DFF);
        k_pack<<<blocks_for(DFF * 256), 256, 0, stream>>>(W_f2   + (size_t)li * DFF * 256, wp_f2, DFF, 256);

        // q = x + pos (bf16), x (bf16)
        k_prep<<<blocks_for(MD), 256, 0, stream>>>(xbuf, posbuf, q_bf, x_bf);

        // GEMM wave counts: (M/32) * (N/64) waves, 32 threads each
        long long w256 = (long long)(M_TOK / 32) * (256 / 64) * 32;
        long long w128 = (long long)(M_TOK / 32) * (128 / 64) * 32;
        long long wDFF = (long long)(M_TOK / 32) * (DFF / 64) * 32;

        k_gemm<<<blocks_for(w256), 256, 0, stream>>>(q_bf, wp_off, b_off + (size_t)li * 256,
                                                     offb, (__bf16*)nullptr, M_TOK, 256, 256, 0);
        k_gemm<<<blocks_for(w128), 256, 0, stream>>>(q_bf, wp_att, b_attn + (size_t)li * 128,
                                                     attnb, (__bf16*)nullptr, M_TOK, 128, 256, 0);
        k_gemm<<<blocks_for(w256), 256, 0, stream>>>(x_bf, wp_val, b_val + (size_t)li * 256,
                                                     valb, (__bf16*)nullptr, M_TOK, 256, 256, 0);

        // deformable sampling -> samp (bf16)   [samp aliases q_bf]
        k_sample<<<blocks_for((long long)M_TOK * NHEADS * 32), 256, 0, stream>>>(
            offb, attnb, valb, samp);

        // out projection -> proj (f32)         [proj aliases offb]
        k_gemm<<<blocks_for(w256), 256, 0, stream>>>(samp, wp_out, b_out + (size_t)li * 256,
                                                     proj, (__bf16*)nullptr, M_TOK, 256, 256, 0);

        // x = LN(x + proj); also bf16 copy for FFN input
        k_addln<<<blocks_for((long long)M_TOK * 32), 256, 0, stream>>>(
            xbuf, proj, ln1_g + (size_t)li * 256, ln1_b + (size_t)li * 256, xbuf, x_bf);

        // FFN1 (+ReLU, bf16 out) and FFN2 (f32)  [ffn2 aliases valb]
        k_gemm<<<blocks_for(wDFF), 256, 0, stream>>>(x_bf, wp_f1, b_f1 + (size_t)li * DFF,
                                                     (float*)nullptr, ffn1b, M_TOK, DFF, 256, 1);
        k_gemm<<<blocks_for(w256), 256, 0, stream>>>(ffn1b, wp_f2, b_f2 + (size_t)li * 256,
                                                     ffn2, (__bf16*)nullptr, M_TOK, 256, DFF, 0);

        // x = LN(x + ffn2)
        k_addln<<<blocks_for((long long)M_TOK * 32), 256, 0, stream>>>(
            xbuf, ffn2, ln2_g + (size_t)li * 256, ln2_b + (size_t)li * 256, xbuf, (__bf16*)nullptr);
    }

    hipMemcpyAsync(d_out, xbuf, MD * sizeof(float), hipMemcpyDeviceToDevice, stream);
}
